// SparseAttention_42030549958781
// MI455X (gfx1250) — matrix operand order repaired and checked by reading, not compiled, not run on hardware
//
#include <hip/hip_runtime.h>

// ---------------------------------------------------------------------------
// entmax-1.5 (alpha-entmax) attention for MI455X / gfx1250 (wave32, WMMA).
// Shapes fixed by reference: B=4, H=16, S=2048, D=64.
// d_out = [ out : 4*16*2048*64 f32 | p_attn : 4*16*2048*2048 f32 ]
//
// Phase 1: scores = scale * Q K^T  via v_wmma_f32_16x16x32_f16 (f32 acc)
// Phase 2: entmax bisection, one wave per score row.  Support compaction:
//          elements with Xa <= tau_lo(init) are exactly 0 for all later taus
//          (tau is monotone non-decreasing), so bisect only over survivors.
// Phase 3: out = P V via WMMA, 4-way k-split reduced through LDS.
// ---------------------------------------------------------------------------

typedef __attribute__((ext_vector_type(16))) _Float16 v16h;
typedef __attribute__((ext_vector_type(8)))  float    v8f;

#define S_LEN   2048
#define D_DIM   64
#define BH_CNT  64
#define QT      16
#define NWAVES  16
#define THREADS 512
#define ST      2052        // padded LDS row stride (floats); 2052 % 64 == 4
#define CAP     512         // compacted-support capacity per row
#define SC_FLOATS   (QT * ST)
#define PART_FLOATS (4 * 16 * 64)
#define SHMEM_BYTES ((SC_FLOATS + QT * CAP + PART_FLOATS) * sizeof(float) + \
                     QT * CAP * sizeof(unsigned short))
#define N_ITERS 50
#define OUT0    ((size_t)BH_CNT * S_LEN * D_DIM)
#define SENTINEL (-1.0e30f)

__device__ __forceinline__ float wred_sum(float v) {
  v += __shfl_xor(v, 1, 32);
  v += __shfl_xor(v, 2, 32);
  v += __shfl_xor(v, 4, 32);
  v += __shfl_xor(v, 8, 32);
  v += __shfl_xor(v, 16, 32);
  return v;
}
__device__ __forceinline__ float wred_max(float v) {
  v = fmaxf(v, __shfl_xor(v, 1, 32));
  v = fmaxf(v, __shfl_xor(v, 2, 32));
  v = fmaxf(v, __shfl_xor(v, 4, 32));
  v = fmaxf(v, __shfl_xor(v, 8, 32));
  v = fmaxf(v, __shfl_xor(v, 16, 32));
  return v;
}

__device__ __forceinline__ void pack8(v16h& v, int off, float4 a, float4 b) {
  v[off + 0] = (_Float16)a.x; v[off + 1] = (_Float16)a.y;
  v[off + 2] = (_Float16)a.z; v[off + 3] = (_Float16)a.w;
  v[off + 4] = (_Float16)b.x; v[off + 5] = (_Float16)b.y;
  v[off + 6] = (_Float16)b.z; v[off + 7] = (_Float16)b.w;
}

template <bool SQ>
__device__ __forceinline__ float pf(float t, float inv) {
  t = fmaxf(t, 0.0f);
  return SQ ? (t * t) : __powf(t, inv);
}

// ---- compact-support bisection: NE register elements per lane -------------
template <int NE, bool SQ>
__device__ __forceinline__ void compact_solve(float* valsR,
                                              const unsigned short* colsR,
                                              float* row, float* pg, int l,
                                              int na, float f_lo, float tau,
                                              float dm, float inv) {
  // sentinel-pad [na, NE*32) so every lane slot is valid
  for (int idx = na + l; idx < NE * 32; idx += 32) valsR[idx] = SENTINEL;
  float r[NE];
#pragma unroll
  for (int i = 0; i < NE; ++i) r[i] = valsR[l + 32 * i];

#pragma unroll 1
  for (int it = 0; it < N_ITERS; ++it) {
    dm *= 0.5f;
    const float tm = tau + dm;
    float sm = 0.0f;
#pragma unroll
    for (int i = 0; i < NE; ++i) sm += pf<SQ>(r[i] - tm, inv);
    const float fm = wred_sum(sm) - 1.0f;   // wave-uniform
    if (fm * f_lo >= 0.0f) tau = tm;
  }

  float pv[NE];
  float ssum = 0.0f;
#pragma unroll
  for (int i = 0; i < NE; ++i) { pv[i] = pf<SQ>(r[i] - tau, inv); ssum += pv[i]; }
  ssum = wred_sum(ssum);
  const float rs = 1.0f / ssum;

  // zero-fill dense row (LDS + global), then scatter the support
  const float4 z4 = {0.0f, 0.0f, 0.0f, 0.0f};
#pragma unroll
  for (int j = 0; j < 16; ++j) {
    *(float4*)(row + j * 128 + l * 4) = z4;
    *(float4*)(pg  + j * 128 + l * 4) = z4;
  }
#pragma unroll
  for (int i = 0; i < NE; ++i) {
    const int idx = l + 32 * i;
    if (idx < na) {
      const int col = colsR[idx];
      const float p = pv[i] * rs;
      row[col] = p;
      pg[col]  = p;
    }
  }
}

// ---- dense fallback (support > CAP; correctness path) ---------------------
template <bool SQ>
__device__ __forceinline__ void entmax_dense(float4* xa, float f_lo, float tau,
                                             float dm, float inv) {
#pragma unroll 1
  for (int it = 0; it < N_ITERS; ++it) {
    dm *= 0.5f;
    const float tm = tau + dm;
    float sm = 0.0f;
#pragma unroll
    for (int j = 0; j < 16; ++j)
      sm += pf<SQ>(xa[j].x - tm, inv) + pf<SQ>(xa[j].y - tm, inv) +
            pf<SQ>(xa[j].z - tm, inv) + pf<SQ>(xa[j].w - tm, inv);
    const float fm = wred_sum(sm) - 1.0f;
    if (fm * f_lo >= 0.0f) tau = tm;
  }
  float ssum = 0.0f;
#pragma unroll
  for (int j = 0; j < 16; ++j) {
    xa[j].x = pf<SQ>(xa[j].x - tau, inv);
    xa[j].y = pf<SQ>(xa[j].y - tau, inv);
    xa[j].z = pf<SQ>(xa[j].z - tau, inv);
    xa[j].w = pf<SQ>(xa[j].w - tau, inv);
    ssum += xa[j].x + xa[j].y + xa[j].z + xa[j].w;
  }
  ssum = wred_sum(ssum);
  const float rs = 1.0f / ssum;
#pragma unroll
  for (int j = 0; j < 16; ++j) {
    xa[j].x *= rs; xa[j].y *= rs; xa[j].z *= rs; xa[j].w *= rs;
  }
}

// ---- per-row entmax: load, compact, dispatch ------------------------------
template <bool SQ>
__device__ __forceinline__ void phase2_row(float* row, float* valsR,
                                           unsigned short* colsR, float* pg,
                                           int l, float am1, float inv) {
  float4 xa[16];
#pragma unroll
  for (int j = 0; j < 16; ++j) {
    float4 v4 = *(const float4*)(row + j * 128 + l * 4);
    xa[j].x = v4.x * am1; xa[j].y = v4.y * am1;
    xa[j].z = v4.z * am1; xa[j].w = v4.w * am1;
  }

  float mx = -3.402823466e38f;
#pragma unroll
  for (int j = 0; j < 16; ++j)
    mx = fmaxf(mx, fmaxf(fmaxf(xa[j].x, xa[j].y), fmaxf(xa[j].z, xa[j].w)));
  mx = wred_max(mx);
  const float tau0 = mx - 1.0f;
  const float dm0  = (mx - __powf(1.0f / (float)S_LEN, am1)) - tau0;

  // compaction (ballot prefix) + f_lo in one pass
  int   base = 0;
  float s    = 0.0f;
  const unsigned ltm = (1u << l) - 1u;
#pragma unroll
  for (int j = 0; j < 16; ++j) {
#pragma unroll
    for (int cc = 0; cc < 4; ++cc) {
      const float v = (cc == 0) ? xa[j].x : (cc == 1) ? xa[j].y
                    : (cc == 2) ? xa[j].z : xa[j].w;
      s += pf<SQ>(v - tau0, inv);
      const bool act = v > tau0;
      const unsigned mask = (unsigned)__ballot(act);
      const int pos = base + __popc(mask & ltm);
      if (act && pos < CAP) {
        valsR[pos] = v;
        colsR[pos] = (unsigned short)(j * 128 + l * 4 + cc);
      }
      base += __popc(mask);
    }
  }
  const float f_lo = wred_sum(s) - 1.0f;
  const int   na   = base;                       // wave-uniform

  if (na <= 128) {
    compact_solve<4, SQ>(valsR, colsR, row, pg, l, na, f_lo, tau0, dm0, inv);
  } else if (na <= CAP) {
    compact_solve<16, SQ>(valsR, colsR, row, pg, l, na, f_lo, tau0, dm0, inv);
  } else {
    entmax_dense<SQ>(xa, f_lo, tau0, dm0, inv);
#pragma unroll
    for (int j = 0; j < 16; ++j) {
      *(float4*)(row + j * 128 + l * 4) = xa[j];
      *(float4*)(pg  + j * 128 + l * 4) = xa[j];
    }
  }
}

__global__ __launch_bounds__(THREADS)
void entmax_attn_kernel(const float* __restrict__ Q,
                        const float* __restrict__ Kmat,
                        const float* __restrict__ V,
                        const float* __restrict__ alphap,
                        float* __restrict__ out) {
  extern __shared__ float smem[];
  float*          sc   = smem;                         // [QT][ST]
  float*          vals = smem + SC_FLOATS;             // [QT][CAP]
  float*          part = vals + QT * CAP;              // [4][16][64]
  unsigned short* cols = (unsigned short*)(part + PART_FLOATS); // [QT][CAP]

  const int tid = threadIdx.x;
  const int wv  = tid >> 5;
  const int l   = tid & 31;
  const int hi  = l >> 4;
  const int ln  = l & 15;

  const int    blk  = blockIdx.x;
  const int    bh   = blk >> 7;
  const int    qt   = blk & 127;
  const size_t base = (size_t)bh * S_LEN * D_DIM;
  const int    q0   = qt * QT;
  const float  scale = 0.125f;       // 1/sqrt(64)

  // ---------------- Phase 1: scores via WMMA -------------------------------
  v16h a_lo, a_hi;
  {
    const float* qr = Q + base + (size_t)(q0 + ln) * D_DIM;
    float4 c0 = *(const float4*)(qr + hi * 8);
    float4 c1 = *(const float4*)(qr + hi * 8 + 4);
    float4 c2 = *(const float4*)(qr + 16 + hi * 8);
    float4 c3 = *(const float4*)(qr + 16 + hi * 8 + 4);
    pack8(a_lo, 0, c0, c1); pack8(a_lo, 8, c2, c3);
    c0 = *(const float4*)(qr + 32 + hi * 8);
    c1 = *(const float4*)(qr + 32 + hi * 8 + 4);
    c2 = *(const float4*)(qr + 48 + hi * 8);
    c3 = *(const float4*)(qr + 48 + hi * 8 + 4);
    pack8(a_hi, 0, c0, c1); pack8(a_hi, 8, c2, c3);
  }

  for (int kt = wv; kt < S_LEN / 16; kt += NWAVES) {
    const float* kr = Kmat + base + (size_t)(kt * 16 + ln) * D_DIM;
    if (kt + NWAVES < S_LEN / 16) {       // prefetch next K tile row (L2)
      const float* nk = kr + (size_t)NWAVES * 16 * D_DIM;
      __builtin_prefetch(nk, 0, 1);
      __builtin_prefetch(nk + 32, 0, 1);
    }
    v16h b_lo, b_hi;
    {
      float4 c0 = *(const float4*)(kr + hi * 8);
      float4 c1 = *(const float4*)(kr + hi * 8 + 4);
      float4 c2 = *(const float4*)(kr + 16 + hi * 8);
      float4 c3 = *(const float4*)(kr + 16 + hi * 8 + 4);
      pack8(b_lo, 0, c0, c1); pack8(b_lo, 8, c2, c3);
      c0 = *(const float4*)(kr + 32 + hi * 8);
      c1 = *(const float4*)(kr + 32 + hi * 8 + 4);
      c2 = *(const float4*)(kr + 48 + hi * 8);
      c3 = *(const float4*)(kr + 48 + hi * 8 + 4);
      pack8(b_hi, 0, c0, c1); pack8(b_hi, 8, c2, c3);
    }
    v8f c = {};
    c = __builtin_amdgcn_wmma_f32_16x16x32_f16(false, a_lo, false, b_lo,
                                               (short)0, c, false, false);
    c = __builtin_amdgcn_wmma_f32_16x16x32_f16(false, a_hi, false, b_hi,
                                               (short)0, c, false, false);
    float* dst = sc + (size_t)(hi * 8) * ST + kt * 16 + ln;
#pragma unroll
    for (int r = 0; r < 8; ++r) dst[r * ST] = c[r] * scale;
  }
  __syncthreads();

  // ---------------- Phase 2: entmax (one wave per row) ---------------------
  {
    const float am1 = alphap[0] - 1.0f;
    const float inv = 1.0f / am1;
    float*          row   = sc + (size_t)wv * ST;
    float*          valsR = vals + (size_t)wv * CAP;
    unsigned short* colsR = cols + (size_t)wv * CAP;
    float* pg = out + OUT0 + ((size_t)bh * S_LEN + (q0 + wv)) * S_LEN;
    if (inv == 2.0f) phase2_row<true >(row, valsR, colsR, pg, l, am1, inv);
    else             phase2_row<false>(row, valsR, colsR, pg, l, am1, inv);
  }
  __syncthreads();

  // ---------------- Phase 3: out = P * V via WMMA --------------------------
  {
    const int nt = wv & 3;
    const int kc = wv >> 2;
    const float* vb = V + base;
    v8f c = {};
    for (int i = 0; i < 16; ++i) {
      const int kb = kc * 512 + i * 32;
      const float* ar = sc + (size_t)ln * ST + kb;
      float4 c0 = *(const float4*)(ar + hi * 16);
      float4 c1 = *(const float4*)(ar + hi * 16 + 4);
      float4 c2 = *(const float4*)(ar + hi * 16 + 8);
      float4 c3 = *(const float4*)(ar + hi * 16 + 12);
      v16h a; pack8(a, 0, c0, c1); pack8(a, 8, c2, c3);
      const int n = nt * 16 + ln;
      const float* vc = vb + (size_t)(kb + hi * 16) * D_DIM + n;
      v16h b;
#pragma unroll
      for (int e = 0; e < 16; ++e) b[e] = (_Float16)vc[e * D_DIM];
      c = __builtin_amdgcn_wmma_f32_16x16x32_f16(false, a, false, b,
                                                 (short)0, c, false, false);
    }
    float* pp = part + (size_t)(kc * 16 + hi * 8) * 64 + nt * 16 + ln;
#pragma unroll
    for (int r = 0; r < 8; ++r) pp[r * 64] = c[r];
  }
  __syncthreads();

  for (int oi = tid; oi < QT * D_DIM; oi += THREADS) {
    const int m = oi >> 6, dcol = oi & 63;
    float s = part[(0 * 16 + m) * 64 + dcol] + part[(1 * 16 + m) * 64 + dcol] +
              part[(2 * 16 + m) * 64 + dcol] + part[(3 * 16 + m) * 64 + dcol];
    out[((size_t)bh * S_LEN + (q0 + m)) * D_DIM + dcol] = s;
  }
}

extern "C" void kernel_launch(void* const* d_in, const int* in_sizes, int n_in,
                              void* d_out, int out_size, void* d_ws, size_t ws_size,
                              hipStream_t stream) {
  (void)in_sizes; (void)n_in; (void)d_ws; (void)ws_size; (void)out_size;
  const float* Q = (const float*)d_in[0];
  const float* K = (const float*)d_in[1];
  const float* V = (const float*)d_in[2];
  const float* A = (const float*)d_in[3];
  float* out = (float*)d_out;

  hipFuncSetAttribute((const void*)entmax_attn_kernel,
                      hipFuncAttributeMaxDynamicSharedMemorySize,
                      (int)SHMEM_BYTES);

  dim3 grid(BH_CNT * (S_LEN / QT));   // 8192 workgroups
  dim3 block(THREADS);                // 16 waves (wave32)
  entmax_attn_kernel<<<grid, block, SHMEM_BYTES, stream>>>(Q, K, V, A, out);
}
